// QuantumFeatureMap_66889820668524
// MI455X (gfx1250) — matrix operand order repaired and checked by reading, not compiled, not run on hardware
//
#include <hip/hip_runtime.h>
#include <hip/hip_bf16.h>

#define N_QUBITS 14
#define N_LAYERS 3
#define DIM (1 << N_QUBITS)
#define BATCH 1024
#define SIM_BLK 256
#define STAGE_KT 4   // k-tiles (of 32) per TDM stage in big GEMM

typedef __attribute__((ext_vector_type(16))) __bf16 v16bf;
typedef __attribute__((ext_vector_type(8)))  float  v8f;
typedef __attribute__((ext_vector_type(4)))  unsigned int u32x4;
typedef __attribute__((ext_vector_type(8)))  int i32x8;
typedef __attribute__((ext_vector_type(4)))  int i32x4;

// ---------------------------------------------------------------------------
// enc = x (1024x64) @ encoding_params (64x14)   -> enc (1024x14) f32
// ---------------------------------------------------------------------------
__global__ void enc_kernel(const float* __restrict__ x,
                           const float* __restrict__ ep,
                           float* __restrict__ enc) {
    int idx = blockIdx.x * blockDim.x + threadIdx.x;
    if (idx >= BATCH * N_QUBITS) return;
    int b = idx / N_QUBITS;
    int q = idx - b * N_QUBITS;
    float s = 0.f;
    #pragma unroll 8
    for (int f = 0; f < 64; ++f)
        s += x[b * 64 + f] * ep[f * N_QUBITS + q];
    enc[idx] = s;
}

// ---------------------------------------------------------------------------
// Pack row-major f32 weight (K x N) into bf16 WMMA B-operand layout.
// B tile (32 x 16): lane half h holds K = 32*t + 16*(j>>3) + 8*h + (j&7)
// packed index e = ((t*N + c)*2 + h)*16 + j
// ---------------------------------------------------------------------------
__global__ void packB_kernel(const float* __restrict__ w,
                             __bf16* __restrict__ out, int N, int total) {
    int e = blockIdx.x * blockDim.x + threadIdx.x;
    if (e >= total) return;
    int j = e & 15;
    int h = (e >> 4) & 1;
    int c = (e >> 5) % N;
    int t = e / (N * 32);
    int k = 32 * t + 16 * (j >> 3) + 8 * h + (j & 7);
    out[e] = (__bf16)w[(size_t)k * N + c];
}

// ---------------------------------------------------------------------------
// Gate helpers (operate on one complex amplitude pair held in registers)
// ---------------------------------------------------------------------------
__device__ __forceinline__ int ins_bit(int x, int q) {
    return ((x >> q) << (q + 1)) | (x & ((1 << q) - 1));
}
// fused RX(ax)*RY(ay)*RZ(az) applied to pair (s0, s1) along one qubit
__device__ __forceinline__ void rot_pair(float cx, float sx, float cy, float sy,
                                         float cz, float sz,
                                         float& r0, float& m0, float& r1, float& m1) {
    float tr0 = cx * r0 + sx * m1, tm0 = cx * m0 - sx * r1;   // RX
    float tr1 = cx * r1 + sx * m0, tm1 = cx * m1 - sx * r0;
    float ur0 = cy * tr0 - sy * tr1, um0 = cy * tm0 - sy * tm1; // RY
    float ur1 = sy * tr0 + cy * tr1, um1 = sy * tm0 + cy * tm1;
    r0 = ur0 * cz + um0 * sz;  m0 = um0 * cz - ur0 * sz;        // RZ
    r1 = ur1 * cz - um1 * sz;  m1 = um1 * cz + ur1 * sz;
}
__device__ __forceinline__ void ry_pair(float c, float s,
                                        float& r0, float& m0, float& r1, float& m1) {
    float a = r0, bb = m0;
    r0 = c * a - s * r1;   m0 = c * bb - s * m1;
    r1 = s * a + c * r1;   m1 = s * bb + c * m1;
}
// pcnot active part: s0' = ca*s0 + i*sa*s1 ; s1' = ca*s1 + i*sa*s0
__device__ __forceinline__ void pc_pair(float ca, float sa,
                                        float& r0, float& m0, float& r1, float& m1) {
    float a = r0, bb = m0;
    r0 = ca * a  - sa * m1;  m0 = ca * bb + sa * r1;
    r1 = ca * r1 - sa * bb;  m1 = ca * m1 + sa * a;
}

// ---------------------------------------------------------------------------
// Statevector simulator: one workgroup per batch element; full 14-qubit state
// (16384 complex) resident in 128 KB LDS (CDNA5 WGP: 320 KB).
// Gates fused: qubit-PAIR passes (4-amp register groups) for encoding and
// RxRyRz layers; entanglers fused pairwise (3-qubit / 8-amp groups).
// 49 LDS passes total instead of 95.
// Output: |amp|^2 as bf16 scattered into WMMA A-operand packed layout:
//   k = 32*t + 16*(j>>3) + 8*h + (j&7), elem = ((t*M + row)*2 + h)*16 + j
// ---------------------------------------------------------------------------
__global__ void __launch_bounds__(SIM_BLK)
sim_kernel(const float* __restrict__ enc,   // 1024 x 14
           const float* __restrict__ rot,   // 3 x 14 x 3
           const float* __restrict__ ent,   // 3 x 13
           __bf16* __restrict__ Ap) {       // packed feats (1024 x 16384)
    extern __shared__ float smem[];
    float* RE = smem;
    float* IMv = smem + DIM;
    const int tid = threadIdx.x;
    const int b = blockIdx.x;

    for (int i = tid; i < DIM; i += SIM_BLK) {
        RE[i] = (i == 0) ? 1.f : 0.f;
        IMv[i] = 0.f;
    }
    __syncthreads();

    // ----- encoding: RY per qubit, two qubits per LDS pass -----
    for (int qp = 0; qp < N_QUBITS; qp += 2) {
        const int q0 = qp, q1 = qp + 1;
        float a0 = enc[b * N_QUBITS + q0] * 0.5f;
        float a1 = enc[b * N_QUBITS + q1] * 0.5f;
        float c0 = cosf(a0), s0 = sinf(a0), c1 = cosf(a1), s1 = sinf(a1);
        for (int p = tid; p < DIM / 4; p += SIM_BLK) {
            int base = ins_bit(ins_bit(p, q0), q1);
            int i00 = base, i01 = base | (1 << q0);
            int i10 = base | (1 << q1), i11 = i01 | (1 << q1);
            float r00 = RE[i00], m00 = IMv[i00], r01 = RE[i01], m01 = IMv[i01];
            float r10 = RE[i10], m10 = IMv[i10], r11 = RE[i11], m11 = IMv[i11];
            ry_pair(c0, s0, r00, m00, r01, m01);   // qubit q0 pairs
            ry_pair(c0, s0, r10, m10, r11, m11);
            ry_pair(c1, s1, r00, m00, r10, m10);   // qubit q1 pairs
            ry_pair(c1, s1, r01, m01, r11, m11);
            RE[i00] = r00; IMv[i00] = m00;  RE[i01] = r01; IMv[i01] = m01;
            RE[i10] = r10; IMv[i10] = m10;  RE[i11] = r11; IMv[i11] = m11;
        }
        __syncthreads();
    }

    // ----- variational layers -----
    for (int l = 0; l < N_LAYERS; ++l) {
        // fused RxRyRz, two qubits per LDS pass
        for (int qp = 0; qp < N_QUBITS; qp += 2) {
            const int q0 = qp, q1 = qp + 1;
            const float* rp0 = rot + ((l * N_QUBITS + q0) * 3);
            const float* rp1 = rot + ((l * N_QUBITS + q1) * 3);
            float cx0 = cosf(rp0[0] * 0.5f), sx0 = sinf(rp0[0] * 0.5f);
            float cy0 = cosf(rp0[1] * 0.5f), sy0 = sinf(rp0[1] * 0.5f);
            float cz0 = cosf(rp0[2] * 0.5f), sz0 = sinf(rp0[2] * 0.5f);
            float cx1 = cosf(rp1[0] * 0.5f), sx1 = sinf(rp1[0] * 0.5f);
            float cy1 = cosf(rp1[1] * 0.5f), sy1 = sinf(rp1[1] * 0.5f);
            float cz1 = cosf(rp1[2] * 0.5f), sz1 = sinf(rp1[2] * 0.5f);
            for (int p = tid; p < DIM / 4; p += SIM_BLK) {
                int base = ins_bit(ins_bit(p, q0), q1);
                int i00 = base, i01 = base | (1 << q0);
                int i10 = base | (1 << q1), i11 = i01 | (1 << q1);
                float r00 = RE[i00], m00 = IMv[i00], r01 = RE[i01], m01 = IMv[i01];
                float r10 = RE[i10], m10 = IMv[i10], r11 = RE[i11], m11 = IMv[i11];
                rot_pair(cx0, sx0, cy0, sy0, cz0, sz0, r00, m00, r01, m01);
                rot_pair(cx0, sx0, cy0, sy0, cz0, sz0, r10, m10, r11, m11);
                rot_pair(cx1, sx1, cy1, sy1, cz1, sz1, r00, m00, r10, m10);
                rot_pair(cx1, sx1, cy1, sy1, cz1, sz1, r01, m01, r11, m11);
                RE[i00] = r00; IMv[i00] = m00;  RE[i01] = r01; IMv[i01] = m01;
                RE[i10] = r10; IMv[i10] = m10;  RE[i11] = r11; IMv[i11] = m11;
            }
            __syncthreads();
        }
        // entanglers: gates g (ctrl g, tgt g+1). Fuse pairs (g,g+1) into one
        // 3-qubit pass over qubits (g, g+1, g+2). Gates 0..11 -> 6 passes,
        // gate 12 single, gate 13 has angle 0 (identity, skipped).
        for (int g = 0; g + 1 < N_QUBITS - 1; g += 2) {
            const int qa = g, qb = g + 1, qc = g + 2;
            float aa = ent[l * (N_QUBITS - 1) + g];
            float ab = ent[l * (N_QUBITS - 1) + g + 1];
            float ca1 = cosf(aa), sa1 = sinf(aa);
            float ca2 = cosf(ab), sa2 = sinf(ab);
            for (int p = tid; p < DIM / 8; p += SIM_BLK) {
                int base = ins_bit(ins_bit(ins_bit(p, qa), qb), qc);
                int idx[8];
                #pragma unroll
                for (int m = 0; m < 8; ++m)
                    idx[m] = base | ((m & 1) << qa) | (((m >> 1) & 1) << qb)
                                  | (((m >> 2) & 1) << qc);
                float r[8], im[8];
                #pragma unroll
                for (int m = 0; m < 8; ++m) { r[m] = RE[idx[m]]; im[m] = IMv[idx[m]]; }
                // gate g: ctrl bit qa==1, target qb: pairs (001,011),(101,111)
                pc_pair(ca1, sa1, r[1], im[1], r[3], im[3]);
                pc_pair(ca1, sa1, r[5], im[5], r[7], im[7]);
                // gate g+1: ctrl bit qb==1, target qc: pairs (010,110),(011,111)
                pc_pair(ca2, sa2, r[2], im[2], r[6], im[6]);
                pc_pair(ca2, sa2, r[3], im[3], r[7], im[7]);
                #pragma unroll
                for (int m = 0; m < 8; ++m) { RE[idx[m]] = r[m]; IMv[idx[m]] = im[m]; }
            }
            __syncthreads();
        }
        { // leftover entangler g = 12: ctrl 12, tgt 13
            const int ctrl = 12, tgt = 13;
            float a = ent[l * (N_QUBITS - 1) + 12];
            float ca = cosf(a), sa = sinf(a);
            for (int p = tid; p < DIM / 2; p += SIM_BLK) {
                int i0 = ((p >> tgt) << (tgt + 1)) | (p & ((1 << tgt) - 1));
                if ((i0 >> ctrl) & 1) {
                    int i1 = i0 | (1 << tgt);
                    float r0 = RE[i0], m0 = IMv[i0], r1 = RE[i1], m1 = IMv[i1];
                    pc_pair(ca, sa, r0, m0, r1, m1);
                    RE[i0] = r0; IMv[i0] = m0; RE[i1] = r1; IMv[i1] = m1;
                }
            }
            __syncthreads();
        }
    }

    // ----- |amp|^2 -> bf16, scattered into WMMA A-operand packed layout -----
    for (int k = tid; k < DIM; k += SIM_BLK) {
        float v = RE[k] * RE[k] + IMv[k] * IMv[k];
        int t = k >> 5, kin = k & 31;
        int jhi = kin >> 4, rem = kin & 15;
        int h = rem >> 3, j = (jhi << 3) | (rem & 7);
        Ap[(size_t)(((size_t)t * BATCH + b) * 2 + h) * 16 + j] = (__bf16)v;
    }
}

// ---------------------------------------------------------------------------
// TDM issue helper (arity differs between toolchains; see CDNA5_HIP.md)
// ---------------------------------------------------------------------------
__device__ __forceinline__ void tdm_load_2d(unsigned int ldsOff,
                                            unsigned long long ga,
                                            int elemsPerChunk, int nChunks,
                                            int chunkStrideElems) {
    u32x4 g0;
    g0.x = 1u;                                       // count = 1 (valid D#)
    g0.y = ldsOff;                                   // lds_addr (bytes)
    g0.z = (unsigned int)(ga & 0xFFFFFFFFull);       // global_addr[31:0]
    g0.w = (unsigned int)((ga >> 32) & 0x1FFFFFFull) // global_addr[56:32]
           | (2u << 30);                             // type = 2 ("image")
    i32x8 g1;
    g1[0] = 1 << 16;                                 // data_size = 1 (2 bytes)
    g1[1] = elemsPerChunk << 16;                     // tensor_dim0[15:0]
    g1[2] = nChunks << 16;                           // tensor_dim1[15:0]
    g1[3] = elemsPerChunk << 16;                     // tile_dim0
    g1[4] = nChunks;                                 // tile_dim1 (tile_dim2=0)
    g1[5] = chunkStrideElems;                        // tensor_dim0_stride[31:0]
    g1[6] = 0;
    g1[7] = 0;
    i32x4 z4 = {0, 0, 0, 0};
#if defined(__clang_major__) && (__clang_major__ >= 23)
    i32x8 z8 = {0, 0, 0, 0, 0, 0, 0, 0};
    __builtin_amdgcn_tensor_load_to_lds(g0, g1, z4, z4, z8, 0);
#else
    __builtin_amdgcn_tensor_load_to_lds(g0, g1, z4, z4, 0);
#endif
}

// ---------------------------------------------------------------------------
// Big WMMA GEMM with TDM-staged B panel.
// Block = 4 waves = 64x64 macro-tile. Wave w: rows [64*bM + 16w), all 4
// N-tiles of the 64-col panel (A fragment reused 4x). B panel (4 k-tiles x
// 64 cols = 16 KB) double-buffered in LDS, filled by tensor_load_to_lds,
// synchronized with s_wait_tensorcnt + one barrier per stage.
// Output: bias + relu -> bf16 packed-A layout for the next GEMM.
// ---------------------------------------------------------------------------
__global__ void __launch_bounds__(128)
wmma_gemm_tdm(const __bf16* __restrict__ Ap, const __bf16* __restrict__ Bp,
              const float* __restrict__ bias, __bf16* __restrict__ outA,
              int M, int N, int Ktiles) {
    __shared__ __align__(64) __bf16 Bsh[2][STAGE_KT * 64 * 32];  // 2 x 16 KB
    const int wv = threadIdx.x >> 5;
    const int lane = threadIdx.x & 31;
    const int l15 = lane & 15, h = lane >> 4;
    const int blocksN = N >> 6;
    const int bM = blockIdx.x / blocksN;
    const int bN = blockIdx.x % blocksN;
    const int rowBase = bM * 64 + wv * 16;
    const int colBase = bN * 64;

    const __bf16* aPtr = Ap + ((size_t)(rowBase + l15) * 2 + h) * 16;
    const size_t aStride = (size_t)M * 32;           // elems per k-tile

    // byte address of this block's 64-col packed-B panel at k-tile 0
    const unsigned long long bPanel =
        (unsigned long long)(uintptr_t)Bp + (unsigned long long)colBase * 64ull;
    const unsigned long long stageBytes =
        (unsigned long long)STAGE_KT * (unsigned long long)N * 64ull;

    const int stages = Ktiles / STAGE_KT;
    v8f acc0 = {}, acc1 = {}, acc2 = {}, acc3 = {};

    if (wv == 0) {
        tdm_load_2d((unsigned int)(uintptr_t)(void*)&Bsh[0][0],
                    bPanel, 64 * 32, STAGE_KT, N * 32);
    }

    for (int s = 0; s < stages; ++s) {
        if (wv == 0) __builtin_amdgcn_s_wait_tensorcnt(0);
        __syncthreads();   // buf[s&1] ready; prev compute done everywhere
        if (wv == 0 && s + 1 < stages) {
            tdm_load_2d((unsigned int)(uintptr_t)(void*)&Bsh[(s + 1) & 1][0],
                        bPanel + (unsigned long long)(s + 1) * stageBytes,
                        64 * 32, STAGE_KT, N * 32);
        }
        const __bf16* bsh = &Bsh[s & 1][0];
        #pragma unroll
        for (int kk = 0; kk < STAGE_KT; ++kk) {
            v16bf a = *(const v16bf*)aPtr;
            aPtr += aStride;
            const __bf16* bk = bsh + kk * (64 * 32);
            v16bf b0 = *(const v16bf*)(bk + ((size_t)(l15 +  0) * 2 + h) * 16);
            v16bf b1 = *(const v16bf*)(bk + ((size_t)(l15 + 16) * 2 + h) * 16);
            v16bf b2 = *(const v16bf*)(bk + ((size_t)(l15 + 32) * 2 + h) * 16);
            v16bf b3 = *(const v16bf*)(bk + ((size_t)(l15 + 48) * 2 + h) * 16);
            acc0 = __builtin_amdgcn_wmma_f32_16x16x32_bf16(false, a, false, b0, (short)0, acc0, false, false);
            acc1 = __builtin_amdgcn_wmma_f32_16x16x32_bf16(false, a, false, b1, (short)0, acc1, false, false);
            acc2 = __builtin_amdgcn_wmma_f32_16x16x32_bf16(false, a, false, b2, (short)0, acc2, false, false);
            acc3 = __builtin_amdgcn_wmma_f32_16x16x32_bf16(false, a, false, b3, (short)0, acc3, false, false);
        }
    }

    // epilogue: bias + relu, write bf16 packed-A for next GEMM
    v8f accs[4] = {acc0, acc1, acc2, acc3};
    #pragma unroll
    for (int n = 0; n < 4; ++n) {
        #pragma unroll
        for (int r = 0; r < 8; ++r) {
            int row = rowBase + r + (h << 3);
            int col = colBase + n * 16 + l15;
            float v = fmaxf(accs[n][r] + bias[col], 0.f);
            int t = col >> 5, kin = col & 31;
            int jhi = kin >> 4, rem = kin & 15;
            int hh = rem >> 3, j = (jhi << 3) | (rem & 7);
            outA[(size_t)(((size_t)t * BATCH + row) * 2 + hh) * 16 + j] = (__bf16)v;
        }
    }
}

// ---------------------------------------------------------------------------
// Small WMMA GEMM: one 16x16 tile per wave, direct global loads.
// ---------------------------------------------------------------------------
__global__ void __launch_bounds__(128)
wmma_gemm(const __bf16* __restrict__ Ap, const __bf16* __restrict__ Bp,
          const float* __restrict__ bias, float* __restrict__ outF,
          __bf16* __restrict__ outA, int M, int N, int Ktiles, int doRelu) {
    int wave = blockIdx.x * (blockDim.x >> 5) + (threadIdx.x >> 5);
    int lane = threadIdx.x & 31;
    int tilesN = N >> 4;
    int tileM = (wave / tilesN) << 4;
    int tileN = (wave % tilesN) << 4;
    int l15 = lane & 15, h = lane >> 4;

    const __bf16* aPtr = Ap + ((size_t)(tileM + l15) * 2 + h) * 16;
    const __bf16* bPtr = Bp + ((size_t)(tileN + l15) * 2 + h) * 16;
    const size_t aStride = (size_t)M * 32;
    const size_t bStride = (size_t)N * 32;

    v8f acc = {};
    #pragma unroll 4
    for (int t = 0; t < Ktiles; ++t) {
        v16bf a = *(const v16bf*)aPtr;
        v16bf bb = *(const v16bf*)bPtr;
        acc = __builtin_amdgcn_wmma_f32_16x16x32_bf16(
            false, a, false, bb, (short)0, acc, false, false);
        aPtr += aStride;
        bPtr += bStride;
    }

    #pragma unroll
    for (int r = 0; r < 8; ++r) {
        int row = tileM + r + (h << 3);
        int col = tileN + l15;
        float v = acc[r] + bias[col];
        if (doRelu) v = fmaxf(v, 0.f);
        if (outF) {
            outF[(size_t)row * N + col] = v;
        } else {
            int t = col >> 5, kin = col & 31;
            int jhi = kin >> 4, rem = kin & 15;
            int hh = rem >> 3, j = (jhi << 3) | (rem & 7);
            outA[(size_t)(((size_t)t * BATCH + row) * 2 + hh) * 16 + j] = (__bf16)v;
        }
    }
}

// ---------------------------------------------------------------------------
extern "C" void kernel_launch(void* const* d_in, const int* in_sizes, int n_in,
                              void* d_out, int out_size, void* d_ws, size_t ws_size,
                              hipStream_t stream) {
    const float* x   = (const float*)d_in[0];   // 1024 x 64
    const float* ep  = (const float*)d_in[1];   // 64 x 14
    const float* rot = (const float*)d_in[2];   // 3 x 14 x 3
    const float* ent = (const float*)d_in[3];   // 3 x 13
    const float* w1  = (const float*)d_in[4];   // 16384 x 256
    const float* b1  = (const float*)d_in[5];   // 256
    const float* w2  = (const float*)d_in[6];   // 256 x 128
    const float* b2  = (const float*)d_in[7];   // 128
    const float* w3  = (const float*)d_in[8];   // 128 x 64
    const float* b3  = (const float*)d_in[9];   // 64
    float* out = (float*)d_out;                 // 1024 x 64

    // workspace carve-up
    char* ws = (char*)d_ws;
    float*  enc = (float*)ws;                      ws += (size_t)BATCH * N_QUBITS * 4;
    __bf16* A1  = (__bf16*)ws;                     ws += (size_t)BATCH * DIM * 2;     // 32 MB
    __bf16* B1  = (__bf16*)ws;                     ws += (size_t)DIM * 256 * 2;       // 8 MB
    __bf16* A2  = (__bf16*)ws;                     ws += (size_t)BATCH * 256 * 2;
    __bf16* B2  = (__bf16*)ws;                     ws += (size_t)256 * 128 * 2;
    __bf16* A3  = (__bf16*)ws;                     ws += (size_t)BATCH * 128 * 2;
    __bf16* B3  = (__bf16*)ws;                     ws += (size_t)128 * 64 * 2;

    // 1) encoding angles
    enc_kernel<<<(BATCH * N_QUBITS) / 256, 256, 0, stream>>>(x, ep, enc);

    // 2) weight packs (f32 -> bf16, WMMA B layout)
    packB_kernel<<<(DIM * 256) / 256, 256, 0, stream>>>(w1, B1, 256, DIM * 256);
    packB_kernel<<<(256 * 128) / 256, 256, 0, stream>>>(w2, B2, 128, 256 * 128);
    packB_kernel<<<(128 * 64) / 256, 256, 0, stream>>>(w3, B3, 64, 128 * 64);

    // 3) LDS-resident statevector simulation -> packed bf16 features
    sim_kernel<<<BATCH, SIM_BLK, 2 * DIM * sizeof(float), stream>>>(enc, rot, ent, A1);

    // 4) MLP
    // layer 1 (dominant): (1024 x 16384) @ (16384 x 256), TDM-staged B, relu
    wmma_gemm_tdm<<<(BATCH / 64) * (256 / 64), 128, 0, stream>>>(
        A1, B1, b1, A2, BATCH, 256, DIM / 32);
    // layer 2: (1024 x 256) @ (256 x 128) -> relu -> packed A3
    wmma_gemm<<<(BATCH / 16) * (128 / 16) / 4, 128, 0, stream>>>(
        A2, B2, b2, nullptr, A3, BATCH, 128, 256 / 32, 1);
    // layer 3: (1024 x 128) @ (128 x 64) -> f32 output
    wmma_gemm<<<(BATCH / 16) * (64 / 16) / 4, 128, 0, stream>>>(
        A3, B3, b3, out, nullptr, BATCH, 64, 128 / 32, 0);
}
